// EdgeDecoder_90065464197846
// MI455X (gfx1250) — matrix operand order repaired and checked by reading, not compiled, not run on hardware
//
#include <hip/hip_runtime.h>

#define HIDDEN 128

typedef __bf16 v16bf __attribute__((ext_vector_type(16)));
typedef __bf16 v8bf  __attribute__((ext_vector_type(8)));
typedef __bf16 v4bf  __attribute__((ext_vector_type(4)));
typedef float  v8f   __attribute__((ext_vector_type(8)));

// ---------------------------------------------------------------------------
// Prep A: split-transpose the concatenated first-layer weight.
//   Wcat[k][n] = (n<128) ? W1[k][n] : W1[128+k][n-128]   (k,n feature/output)
//   WhiT[n][k] = bf16(Wcat[k][n]);  WloT[n][k] = bf16(Wcat[k][n] - WhiT)
// Transposed so a WMMA B-fragment (two runs of 8 consecutive K) is 2 contiguous b128s.
// ---------------------------------------------------------------------------
__global__ __launch_bounds__(256) void edgedec_prep_w(
    const float* __restrict__ W1,
    __bf16* __restrict__ whiT,
    __bf16* __restrict__ wloT)
{
  const int tid = blockIdx.x * 256 + threadIdx.x;   // 256*128 threads
  if (tid >= 256 * HIDDEN) return;
  const int n = tid >> 7;        // output column 0..255
  const int k = tid & 127;       // input feature 0..127
  const float w = (n < HIDDEN) ? W1[k * HIDDEN + n]
                               : W1[(HIDDEN + k) * HIDDEN + (n - HIDDEN)];
  const __bf16 h = (__bf16)w;
  whiT[tid] = h;                               // tid == n*128 + k
  wloT[tid] = (__bf16)(w - (float)h);
}

// ---------------------------------------------------------------------------
// Prep B: split z into bf16 hi/lo tables (row-major, same layout as z).
// 4 elements per thread, vector loads/stores.
// ---------------------------------------------------------------------------
__global__ __launch_bounds__(256) void edgedec_prep_z(
    const float* __restrict__ z,
    __bf16* __restrict__ zhi,
    __bf16* __restrict__ zlo,
    int total4)                                  // total elements / 4
{
  const int tid = blockIdx.x * 256 + threadIdx.x;
  if (tid >= total4) return;
  const float4 x = ((const float4*)z)[tid];
  v4bf h, l;
  h[0] = (__bf16)x.x; l[0] = (__bf16)(x.x - (float)h[0]);
  h[1] = (__bf16)x.y; l[1] = (__bf16)(x.y - (float)h[1]);
  h[2] = (__bf16)x.z; l[2] = (__bf16)(x.z - (float)h[2]);
  h[3] = (__bf16)x.w; l[3] = (__bf16)(x.w - (float)h[3]);
  ((v4bf*)zhi)[tid] = h;
  ((v4bf*)zlo)[tid] = l;
}

// ---------------------------------------------------------------------------
// Stage 1: AB[node, 0:128]   = z[node] @ W1[0:128, :]
//          AB[node, 128:256] = z[node] @ W1[128:256, :]
// One wave -> 16(M) x 64(N) tile. Split-product bf16 WMMA, f32 accumulate:
//   x*w ~= xhi*whi + xhi*wlo + xlo*whi  (error ~2^-16, near-fp32)
// All fragments are assembled from contiguous b128 loads of pre-split tables.
// ---------------------------------------------------------------------------
__global__ __launch_bounds__(256) void edgedec_stage1(
    const __bf16* __restrict__ zhi,
    const __bf16* __restrict__ zlo,
    const __bf16* __restrict__ whiT,
    const __bf16* __restrict__ wloT,
    float* __restrict__ AB,
    int nNodes, int nRowBlocks)
{
  const int lane = threadIdx.x & 31;
  const int wave = (blockIdx.x * blockDim.x + threadIdx.x) >> 5;
  const int rowBlock = wave >> 2;           // wave-uniform -> full-EXEC early out
  if (rowBlock >= nRowBlocks) return;
  const int colGroup = wave & 3;            // 4 groups of 64 output columns
  const int m0   = rowBlock * 16;
  const int n0   = colGroup * 64;
  const int half = lane >> 4;               // 0: lanes 0-15, 1: lanes 16-31
  const int l15  = lane & 15;

  int mRow = m0 + l15;
  if (mRow >= nNodes) mRow = nNodes - 1;    // clamp (exact for 100000)
  const __bf16* __restrict__ zhr = zhi + (size_t)mRow * HIDDEN;
  const __bf16* __restrict__ zlr = zlo + (size_t)mRow * HIDDEN;

  v8f c[4] = {v8f{}, v8f{}, v8f{}, v8f{}};

  #pragma unroll
  for (int k0 = 0; k0 < HIDDEN; k0 += 32) {
    // A fragment 16x32 bf16: lane half h, elem e -> K = k0 + h*8 + (e<8?e:e+8)
    // => elems 0-7 and 8-15 are two contiguous 16B runs of the z row.
    const int ka0 = k0 + half * 8;
    v8bf ah0 = *(const v8bf*)(zhr + ka0);
    v8bf ah1 = *(const v8bf*)(zhr + ka0 + 16);
    v8bf al0 = *(const v8bf*)(zlr + ka0);
    v8bf al1 = *(const v8bf*)(zlr + ka0 + 16);
    v16bf ahi = __builtin_shufflevector(ah0, ah1, 0,1,2,3,4,5,6,7,8,9,10,11,12,13,14,15);
    v16bf alo = __builtin_shufflevector(al0, al1, 0,1,2,3,4,5,6,7,8,9,10,11,12,13,14,15);

    #pragma unroll
    for (int t = 0; t < 4; ++t) {
      // B fragment 32x16 bf16: lane half h, elem e -> K = k0 + h*8 + (e<8?e:e+8)
      // => two contiguous 8-K runs for column (n0 + t*16 + l15) in transposed table.
      const size_t wb = (size_t)(n0 + t * 16 + l15) * HIDDEN + k0 + half * 8;
      v8bf bh0 = *(const v8bf*)(whiT + wb);
      v8bf bh1 = *(const v8bf*)(whiT + wb + 16);
      v8bf bl0 = *(const v8bf*)(wloT + wb);
      v8bf bl1 = *(const v8bf*)(wloT + wb + 16);
      v16bf bhi = __builtin_shufflevector(bh0, bh1, 0,1,2,3,4,5,6,7,8,9,10,11,12,13,14,15);
      v16bf blo = __builtin_shufflevector(bl0, bl1, 0,1,2,3,4,5,6,7,8,9,10,11,12,13,14,15);

      c[t] = __builtin_amdgcn_wmma_f32_16x16x32_bf16(
                 false, ahi, false, bhi, (short)0, c[t], false, false);
      c[t] = __builtin_amdgcn_wmma_f32_16x16x32_bf16(
                 false, ahi, false, blo, (short)0, c[t], false, false);
      c[t] = __builtin_amdgcn_wmma_f32_16x16x32_bf16(
                 false, alo, false, bhi, (short)0, c[t], false, false);
    }
  }

  // C/D layout: VGPR i -> row m0 + i + 8*half, col = n0 + l15
  #pragma unroll
  for (int i = 0; i < 8; ++i) {
    const int row = m0 + half * 8 + i;
    if (row < nNodes) {
      float* o = AB + (size_t)row * 256 + n0 + l15;
      o[0]  = c[0][i];
      o[16] = c[1][i];
      o[32] = c[2][i];
      o[48] = c[3][i];
    }
  }
}

// ---------------------------------------------------------------------------
// Stage 2: out[e] = ReLU(A[src[e]] + B[dst[e]] + b1) . W2 + b2
// Bottleneck stage: 1.6 GB of coalesced 512B gathers from the L2-resident
// AB table (102 MB << 192 MB L2). One wave per edge-pair; lane owns 4
// contiguous features; wave32 shuffle-xor reduction.
// ---------------------------------------------------------------------------
__global__ __launch_bounds__(256) void edgedec_stage2(
    const int* __restrict__ eidx,
    const float* __restrict__ AB,
    const float* __restrict__ b1,
    const float* __restrict__ W2,
    const float* __restrict__ b2p,
    float* __restrict__ out,
    int E)
{
  const int lane = threadIdx.x & 31;
  const int wid  = (blockIdx.x * blockDim.x + threadIdx.x) >> 5;
  const int nw   = (gridDim.x * blockDim.x) >> 5;
  const int* __restrict__ src = eidx;
  const int* __restrict__ dst = eidx + E;

  const float4 b1v = *(const float4*)(b1 + 4 * lane);
  const float4 w2v = *(const float4*)(W2 + 4 * lane);
  const float  b2  = *b2p;

  for (long long e = 2LL * wid; e < E; e += 2LL * nw) {
    const bool pair = (e + 1 < E);
    const int s0 = src[e];
    const int d0 = dst[e];
    const int s1 = pair ? src[e + 1] : s0;
    const int d1 = pair ? dst[e + 1] : d0;

    const float4 x0 = *(const float4*)(AB + (size_t)s0 * 256 + 4 * lane);
    const float4 y0 = *(const float4*)(AB + (size_t)d0 * 256 + 128 + 4 * lane);
    const float4 x1 = *(const float4*)(AB + (size_t)s1 * 256 + 4 * lane);
    const float4 y1 = *(const float4*)(AB + (size_t)d1 * 256 + 128 + 4 * lane);

    float p0 = 0.f, p1 = 0.f;
    p0 = fmaf(fmaxf(x0.x + y0.x + b1v.x, 0.f), w2v.x, p0);
    p0 = fmaf(fmaxf(x0.y + y0.y + b1v.y, 0.f), w2v.y, p0);
    p0 = fmaf(fmaxf(x0.z + y0.z + b1v.z, 0.f), w2v.z, p0);
    p0 = fmaf(fmaxf(x0.w + y0.w + b1v.w, 0.f), w2v.w, p0);
    p1 = fmaf(fmaxf(x1.x + y1.x + b1v.x, 0.f), w2v.x, p1);
    p1 = fmaf(fmaxf(x1.y + y1.y + b1v.y, 0.f), w2v.y, p1);
    p1 = fmaf(fmaxf(x1.z + y1.z + b1v.z, 0.f), w2v.z, p1);
    p1 = fmaf(fmaxf(x1.w + y1.w + b1v.w, 0.f), w2v.w, p1);

    #pragma unroll
    for (int off = 16; off > 0; off >>= 1) {
      p0 += __shfl_xor(p0, off, 32);
      p1 += __shfl_xor(p1, off, 32);
    }
    if (lane == 0) {
      out[e] = p0 + b2;
      if (pair) out[e + 1] = p1 + b2;
    }
  }
}

// ---------------------------------------------------------------------------
// Inputs (setup_inputs order): z[100000*128] f32, edge_index[2*E] int,
// W1[256*128] f32, b1[128] f32, W2[128] f32, b2[1] f32.
// Workspace layout (16B aligned throughout):
//   AB   : nNodes*256 f32               (~102.4 MB, L2-resident)
//   WhiT : 256*128 bf16   (64 KB)
//   WloT : 256*128 bf16   (64 KB)
//   Zhi  : nNodes*128 bf16 (~25.6 MB)
//   Zlo  : nNodes*128 bf16 (~25.6 MB)
// ---------------------------------------------------------------------------
extern "C" void kernel_launch(void* const* d_in, const int* in_sizes, int n_in,
                              void* d_out, int out_size, void* d_ws, size_t ws_size,
                              hipStream_t stream)
{
  const float* z  = (const float*)d_in[0];
  const int*   ei = (const int*)d_in[1];
  const float* W1 = (const float*)d_in[2];
  const float* b1 = (const float*)d_in[3];
  const float* W2 = (const float*)d_in[4];
  const float* b2 = (const float*)d_in[5];
  float* out = (float*)d_out;

  const int nNodes = in_sizes[0] / HIDDEN;
  const int E      = in_sizes[1] / 2;
  const int zElems = nNodes * HIDDEN;

  char* wsb = (char*)d_ws;
  const size_t abBytes = (size_t)nNodes * 256 * sizeof(float);
  float*  AB   = (float*)wsb;
  __bf16* whiT = (__bf16*)(wsb + abBytes);
  __bf16* wloT = whiT + 256 * HIDDEN;
  __bf16* zhi  = wloT + 256 * HIDDEN;
  __bf16* zlo  = zhi + (size_t)zElems;

  // Prep: split weights (transposed) and z into bf16 hi/lo tables.
  edgedec_prep_w<<<(256 * HIDDEN + 255) / 256, 256, 0, stream>>>(W1, whiT, wloT);
  const int total4 = zElems / 4;
  edgedec_prep_z<<<(total4 + 255) / 256, 256, 0, stream>>>(z, zhi, zlo, total4);

  // Stage 1: dense GEMM [nNodes x 128] @ [128 x 256] via WMMA.
  const int nRowBlocks = (nNodes + 15) / 16;
  const int waves1     = nRowBlocks * 4;     // 4 col-groups of 64
  const int blocks1    = (waves1 + 7) / 8;   // 8 waves (256 thr) per block
  edgedec_stage1<<<blocks1, 256, 0, stream>>>(zhi, zlo, whiT, wloT, AB,
                                              nNodes, nRowBlocks);

  // Stage 2: per-edge gather + fused bias/ReLU/dot.
  edgedec_stage2<<<2048, 256, 0, stream>>>(ei, AB, b1, W2, b2, out, E);
}
